// LE_Attention_31464930410907
// MI455X (gfx1250) — matrix operand order repaired and checked by reading, not compiled, not run on hardware
//
#include <hip/hip_runtime.h>
#include <stdint.h>
#include <stddef.h>

// ---- problem constants (B=4, N=1024, C=1024, H=16, hd=64, NB=63) ----
#define BATCH 4
#define SEQ   1024
#define CDIM  1024
#define NH    16
#define HD    64
#define NB    63

typedef __attribute__((ext_vector_type(16))) __bf16 bf16x16;
typedef __attribute__((ext_vector_type(8)))  __bf16 bf16x8;
typedef __attribute__((ext_vector_type(8)))  float  floatx8;
typedef __attribute__((ext_vector_type(4)))  int    i32x4;

// ---------- bf16 helpers: native casts -> hardware v_cvt ----------
__device__ __forceinline__ __bf16 f2bf(float f) { return (__bf16)f; }
__device__ __forceinline__ float  bf2f(__bf16 b) { return (float)b; }

__device__ __forceinline__ floatx8 wmma_bf16(bf16x16 a, bf16x16 b, floatx8 c) {
  return __builtin_amdgcn_wmma_f32_16x16x32_bf16(false, a, false, b, (short)0, c,
                                                 false, false);
}

// ---------- CDNA5 async global->LDS staging (guarded; fallback = plain copy) ----------
#if defined(__gfx1250__) && __has_builtin(__builtin_amdgcn_global_load_async_to_lds_b128)
#define HAVE_ASYNC_LDS 1
#else
#define HAVE_ASYNC_LDS 0
#endif

#if HAVE_ASYNC_LDS
// builtin expects (int4 __device__*, int4 __shared__*, imm offset, imm cpol)
#define GPTR(p) ((__attribute__((address_space(1))) i32x4*)(p))
#define LPTR(p) ((__attribute__((address_space(3))) i32x4*)(p))
#endif

__device__ __forceinline__ void stage16B(const void* gsrc, void* ldst) {
#if HAVE_ASYNC_LDS
  __builtin_amdgcn_global_load_async_to_lds_b128(GPTR(gsrc), LPTR(ldst), 0, 0);
#else
  *(bf16x8*)ldst = *(const bf16x8*)gsrc;
#endif
}
__device__ __forceinline__ void wait_stage() {
#if HAVE_ASYNC_LDS
#if __has_builtin(__builtin_amdgcn_s_wait_asynccnt)
  __builtin_amdgcn_s_wait_asynccnt(0);
#else
  asm volatile("s_wait_asynccnt 0" ::: "memory");
#endif
#endif
}

// ---------- fragment loaders (CDNA5 wave32 layouts, ISA 7.12.2) ----------
// A 16x32 bf16: lane M = lane%16; half = lane/16; a[0..7]=K(8h..8h+7), a[8..15]=K(16+8h..)
__device__ __forceinline__ bf16x16 frag_a_f32(const float* rowbase, int half) {
  bf16x16 a;
#pragma unroll
  for (int t = 0; t < 8; ++t) a[t] = f2bf(rowbase[8 * half + t]);
#pragma unroll
  for (int t = 0; t < 8; ++t) a[8 + t] = f2bf(rowbase[16 + 8 * half + t]);
  return a;
}
__device__ __forceinline__ bf16x16 frag_a_bf(const __bf16* rowbase, int half) {
  bf16x8 lo = *(const bf16x8*)(rowbase + 8 * half);
  bf16x8 hi = *(const bf16x8*)(rowbase + 16 + 8 * half);
  return __builtin_shufflevector(lo, hi, 0,1,2,3,4,5,6,7,8,9,10,11,12,13,14,15);
}
// B 32x16 bf16: lane N = lane%16; half = lane/16; b[0..7]=K(8h..8h+7), b[8..15]=K(16+8h..)
__device__ __forceinline__ bf16x16 frag_b_f32(const float* colbase, int half) {
  bf16x16 b;
#pragma unroll
  for (int t = 0; t < 8; ++t) b[t] = f2bf(colbase[8 * half + t]);
#pragma unroll
  for (int t = 0; t < 8; ++t) b[8 + t] = f2bf(colbase[16 + 8 * half + t]);
  return b;
}
__device__ __forceinline__ bf16x16 frag_b_bf(const __bf16* colbase, int half) {
  bf16x8 lo = *(const bf16x8*)(colbase + 8 * half);
  bf16x8 hi = *(const bf16x8*)(colbase + 16 + 8 * half);
  return __builtin_shufflevector(lo, hi, 0,1,2,3,4,5,6,7,8,9,10,11,12,13,14,15);
}

// ======================= Kernel 1: QKV projection =======================
// qkv = x @ w_qkv^T ; each wave computes a 16x64 strip (A fragment reused 4x).
// All four B fragments are materialized before the WMMA group to avoid the
// per-WMMA 4-NOP WAR hazard on fragment registers (ISA 7.12.1).
__global__ void __launch_bounds__(128) k_qkv(const float* __restrict__ x,
                                             const float* __restrict__ wqkv,
                                             __bf16* __restrict__ Qb,
                                             __bf16* __restrict__ Kb,
                                             __bf16* __restrict__ Vb) {
  const int wave = threadIdx.x >> 5, lane = threadIdx.x & 31;
  const int nn = lane & 15, half = lane >> 4;
  const int NG = (3 * CDIM) / 64;                 // 48 col groups
  const int wid = blockIdx.x * 4 + wave;
  const int tm = wid / NG, tg = wid % NG;
  const int row0 = tm * 16, col0 = tg * 64;

  const float* arow = x + (size_t)(row0 + nn) * CDIM;
  const float* b0 = wqkv + (size_t)(col0 + nn) * CDIM;

  floatx8 acc0 = {}, acc1 = {}, acc2 = {}, acc3 = {};
#pragma unroll 2
  for (int k0 = 0; k0 < CDIM; k0 += 32) {
    __builtin_prefetch(arow + k0 + 64, 0, 1);
    __builtin_prefetch(b0 + k0 + 64, 0, 1);
    const bf16x16 a  = frag_a_f32(arow + k0, half);
    const bf16x16 w0 = frag_b_f32(b0 + k0, half);
    const bf16x16 w1 = frag_b_f32(b0 + (size_t)16 * CDIM + k0, half);
    const bf16x16 w2 = frag_b_f32(b0 + (size_t)32 * CDIM + k0, half);
    const bf16x16 w3 = frag_b_f32(b0 + (size_t)48 * CDIM + k0, half);
    acc0 = wmma_bf16(a, w0, acc0);
    acc1 = wmma_bf16(a, w1, acc1);
    acc2 = wmma_bf16(a, w2, acc2);
    acc3 = wmma_bf16(a, w3, acc3);
  }
  const int s = col0 >> 10;                       // 0=q 1=k 2=v, uniform per wave
  __bf16* dstBuf = (s == 0) ? Qb : ((s == 1) ? Kb : Vb);
  const float mul = (s == 0) ? 0.125f : 1.0f;     // hd^-0.5 folded into q
  const int h = (col0 & 1023) >> 6;               // uniform per wave
#pragma unroll
  for (int r = 0; r < 8; ++r) {
    const int row = row0 + r + 8 * half;
    const int bb = row >> 10, tok = row & 1023;
    __bf16* dst = dstBuf + ((size_t)(bb * NH + h) * SEQ + tok) * HD;
    dst[nn]      = f2bf(acc0[r] * mul);
    dst[16 + nn] = f2bf(acc1[r] * mul);
    dst[32 + nn] = f2bf(acc2[r] * mul);
    dst[48 + nn] = f2bf(acc3[r] * mul);
  }
}

// =============== Kernel 2: RPE contextual bias panels ===============
// qL[b,h,n,m] = sum_d q[n,d] * rpe_k[h,d,m]     (q already scaled)
// kL[b,h,n,m] = sum_d (k[n,d]*scale) * rpe_q[h,d,m]
__global__ void k_rpe_qk(const __bf16* __restrict__ Qb, const __bf16* __restrict__ Kb,
                         const float* __restrict__ rpeq, const float* __restrict__ rpek,
                         float* __restrict__ qL, float* __restrict__ kL) {
  const size_t id = (size_t)blockIdx.x * blockDim.x + threadIdx.x;  // B*H*N*64
  const int m = (int)(id & 63);
  const size_t bhn = id >> 6;
  const int h = (int)((bhn >> 10) & (NH - 1));
  if (m >= NB) { qL[id] = 0.f; kL[id] = 0.f; return; }
  const __bf16* q = Qb + bhn * HD;
  const __bf16* k = Kb + bhn * HD;
  const float* tq = rpeq + (size_t)h * HD * NB + m;   // [h][d][m]
  const float* tk = rpek + (size_t)h * HD * NB + m;
  float sq = 0.f, sk = 0.f;
#pragma unroll 8
  for (int d = 0; d < HD; ++d) {
    sq += bf2f(q[d]) * tk[(size_t)d * NB];
    sk += bf2f(k[d]) * tq[(size_t)d * NB];
  }
  qL[id] = sq;
  kL[id] = sk * 0.125f;
}

// ======================= Kernel 3: fused attention =======================
// grid = B*H*(N/64); block = 128 (4 waves); each wave owns 16 query rows.
__global__ void __launch_bounds__(128) k_attn(const __bf16* __restrict__ Qb,
                                              const __bf16* __restrict__ Kb,
                                              const __bf16* __restrict__ Vb,
                                              const float* __restrict__ qL,
                                              const float* __restrict__ kL,
                                              const float* __restrict__ rpev,
                                              __bf16* __restrict__ Ob) {
  extern __shared__ float lds[];
  const int wave = threadIdx.x >> 5, lane = threadIdx.x & 31;
  const int nn = lane & 15, half = lane >> 4;
  const int bh = blockIdx.x >> 4;          // b*16 + h
  const int rowblk = blockIdx.x & 15;
  const int h = bh & (NH - 1);
  const int i0 = rowblk * 64 + wave * 16;

  float* sc    = lds + (size_t)wave * 16 * SEQ;                      // 16x1024
  float* Ssum  = lds + 4 * 16 * SEQ + wave * 16 * 64;                // 16x64
  float* rpeo  = lds + 4 * 16 * SEQ + 4 * 16 * 64 + wave * 16 * 64;  // 16x64
  __bf16* kst  = (__bf16*)(lds + 4 * 16 * SEQ + 8 * 16 * 64);        // 32x64 bf16
  __bf16* vstT = kst + 32 * HD;                                      // 64x32 bf16

  // ---- Q fragments (resident for whole kernel) ----
  const __bf16* Qrow = Qb + ((size_t)bh * SEQ + (i0 + nn)) * HD;
  const bf16x16 qa0 = frag_a_bf(Qrow, half);
  const bf16x16 qa1 = frag_a_bf(Qrow + 32, half);

  const float* qLp = qL + (size_t)bh * SEQ * 64;
  const float* kLp = kL + (size_t)bh * SEQ * 64;

  // ---- Phase 1: scores + RPE biases into LDS panel (K staged async, shared) ----
  for (int jt2 = 0; jt2 < SEQ / 32; ++jt2) {
    const int j0 = jt2 * 32;
    __syncthreads();  // previous K tile fully consumed
    {  // stage 32x64 bf16 tile (4 KB): two 16B async transfers per thread
      const char* g = (const char*)(Kb + ((size_t)bh * SEQ + j0) * HD);
      stage16B(g + threadIdx.x * 32, (char*)kst + threadIdx.x * 32);
      stage16B(g + threadIdx.x * 32 + 16, (char*)kst + threadIdx.x * 32 + 16);
    }
    wait_stage();
    __syncthreads();

#pragma unroll
    for (int sub = 0; sub < 2; ++sub) {
      const int jb = sub * 16;
      bf16x16 kb0 = frag_b_bf(kst + (jb + nn) * HD, half);       // d 0..31
      bf16x16 kb1 = frag_b_bf(kst + (jb + nn) * HD + 32, half);  // d 32..63
      floatx8 acc = {};
      acc = wmma_bf16(qa0, kb0, acc);
      acc = wmma_bf16(qa1, kb1, acc);
      const int j = j0 + jb + nn;
#pragma unroll
      for (int r = 0; r < 8; ++r) {
        const int m = r + 8 * half;
        const int i = i0 + m;
        int rel = j - i;
        rel = rel < -31 ? -31 : (rel > 31 ? 31 : rel);
        const int bq = rel + 31;        // bucket for qL term: rp[i,j]
        const int bk = 62 - bq;         // bucket for kL term: rp[j,i]
        sc[m * SEQ + j] = acc[r] + qLp[(size_t)i * 64 + bq] + kLp[(size_t)j * 64 + bk];
      }
    }
  }
  __syncthreads();

  // ---- Phase 2: softmax over 1024 keys; bucket segment-sum scatter ----
  {
    const int row = lane & 15, seg = lane >> 4;  // 2 lanes per row
    float* base = sc + row * SEQ + seg * 512;
    float mx = -3.4e38f;
    for (int t = 0; t < 512; ++t) mx = fmaxf(mx, base[t]);
    mx = fmaxf(mx, __shfl_xor(mx, 16, 32));
    float sum = 0.f;
    for (int t = 0; t < 512; ++t) {
      const float e = __expf(base[t] - mx);
      base[t] = e;
      sum += e;
    }
    sum += __shfl_xor(sum, 16, 32);
    const float inv = 1.f / sum;

    for (int t = lane; t < 16 * 64; t += 32) Ssum[t] = 0.f;  // in-order LDS per wave

    const int i = i0 + row;
    for (int t = 0; t < 512; ++t) {
      const float p = base[t] * inv;
      base[t] = p;                                  // normalized prob back to panel
      const int j = seg * 512 + t;
      int rel = j - i;
      rel = rel < -31 ? -31 : (rel > 31 ? 31 : rel);
      atomicAdd(&Ssum[row * 64 + rel + 31], p);     // ds_add_f32 scatter
    }
  }
  __syncthreads();

  // ---- Phase 2b: rpe_v mini-contraction: rpeo[m,d] = sum_b Ssum[m,b]*Lv[h,b,d] ----
  {
    const float* Lv = rpev + (size_t)h * NB * HD;
    for (int idx = lane; idx < 16 * 64; idx += 32) {
      const int m = idx >> 6, d = idx & 63;
      float a = 0.f;
#pragma unroll 9
      for (int b = 0; b < NB; ++b) a += Ssum[m * 64 + b] * Lv[(size_t)b * HD + d];
      rpeo[idx] = a;
    }
  }

  // ---- Phase 3: out = P @ V  (A from LDS panel, V staged transposed in LDS) ----
  floatx8 o0 = {}, o1 = {}, o2 = {}, o3 = {};
  for (int kc = 0; kc < SEQ / 32; ++kc) {
    __syncthreads();  // previous V chunk consumed / rpeo settled on first iter
    {  // stage rows kc*32..+31 transposed: vstT[d*32 + j] = V[j, d]
      const int j = threadIdx.x >> 2;            // 0..31
      const int d0 = (threadIdx.x & 3) * 16;     // 0,16,32,48
      const __bf16* src = Vb + ((size_t)bh * SEQ + kc * 32 + j) * HD + d0;
      bf16x8 v0 = *(const bf16x8*)(src);
      bf16x8 v1 = *(const bf16x8*)(src + 8);
#pragma unroll
      for (int t = 0; t < 8; ++t) {
        vstT[(d0 + t) * 32 + j] = v0[t];
        vstT[(d0 + 8 + t) * 32 + j] = v1[t];
      }
    }
    __syncthreads();

    const float* prow = sc + nn * SEQ + kc * 32;
    bf16x16 pa;
#pragma unroll
    for (int t = 0; t < 8; ++t) pa[t] = f2bf(prow[8 * half + t]);
#pragma unroll
    for (int t = 0; t < 8; ++t) pa[8 + t] = f2bf(prow[16 + 8 * half + t]);

    // B fragments: two 16B ds loads per lane from transposed stage
    bf16x16 vb0 = frag_b_bf(vstT + (0 * 16 + nn) * 32, half);
    bf16x16 vb1 = frag_b_bf(vstT + (1 * 16 + nn) * 32, half);
    bf16x16 vb2 = frag_b_bf(vstT + (2 * 16 + nn) * 32, half);
    bf16x16 vb3 = frag_b_bf(vstT + (3 * 16 + nn) * 32, half);

    o0 = wmma_bf16(pa, vb0, o0);
    o1 = wmma_bf16(pa, vb1, o1);
    o2 = wmma_bf16(pa, vb2, o2);
    o3 = wmma_bf16(pa, vb3, o3);
  }

  // ---- Epilogue: add rpe_v term, write [B,N,C] bf16 for final projection ----
  const int b = bh >> 4;
#pragma unroll
  for (int r = 0; r < 8; ++r) {
    const int m = r + 8 * half;
    const int i = i0 + m;
    __bf16* dst = Ob + ((size_t)b * SEQ + i) * CDIM + h * HD;
    dst[nn]      = f2bf(o0[r] + rpeo[m * 64 + nn]);
    dst[16 + nn] = f2bf(o1[r] + rpeo[m * 64 + 16 + nn]);
    dst[32 + nn] = f2bf(o2[r] + rpeo[m * 64 + 32 + nn]);
    dst[48 + nn] = f2bf(o3[r] + rpeo[m * 64 + 48 + nn]);
  }
}

// ======================= Kernel 4: output projection =======================
// Each wave computes a 16x64 strip (A fragment reused 4x).
__global__ void __launch_bounds__(128) k_proj(const __bf16* __restrict__ Ob,
                                              const float* __restrict__ wp,
                                              const float* __restrict__ bp,
                                              float* __restrict__ y) {
  const int wave = threadIdx.x >> 5, lane = threadIdx.x & 31;
  const int nn = lane & 15, half = lane >> 4;
  const int NG = CDIM / 64;                       // 16 col groups
  const int wid = blockIdx.x * 4 + wave;
  const int tm = wid / NG, tg = wid % NG;
  const int row0 = tm * 16, col0 = tg * 64;

  const __bf16* arow = Ob + (size_t)(row0 + nn) * CDIM;
  const float* b0 = wp + (size_t)(col0 + nn) * CDIM;

  floatx8 acc0 = {}, acc1 = {}, acc2 = {}, acc3 = {};
#pragma unroll 2
  for (int k0 = 0; k0 < CDIM; k0 += 32) {
    __builtin_prefetch(arow + k0 + 64, 0, 1);
    __builtin_prefetch(b0 + k0 + 64, 0, 1);
    const bf16x16 a  = frag_a_bf(arow + k0, half);
    const bf16x16 w0 = frag_b_f32(b0 + k0, half);
    const bf16x16 w1 = frag_b_f32(b0 + (size_t)16 * CDIM + k0, half);
    const bf16x16 w2 = frag_b_f32(b0 + (size_t)32 * CDIM + k0, half);
    const bf16x16 w3 = frag_b_f32(b0 + (size_t)48 * CDIM + k0, half);
    acc0 = wmma_bf16(a, w0, acc0);
    acc1 = wmma_bf16(a, w1, acc1);
    acc2 = wmma_bf16(a, w2, acc2);
    acc3 = wmma_bf16(a, w3, acc3);
  }
  const float bias0 = bp[col0 + nn];
  const float bias1 = bp[col0 + 16 + nn];
  const float bias2 = bp[col0 + 32 + nn];
  const float bias3 = bp[col0 + 48 + nn];
#pragma unroll
  for (int r = 0; r < 8; ++r) {
    const int row = row0 + r + 8 * half;
    float* dst = y + (size_t)row * CDIM + col0;
    dst[nn]      = acc0[r] + bias0;
    dst[16 + nn] = acc1[r] + bias1;
    dst[32 + nn] = acc2[r] + bias2;
    dst[48 + nn] = acc3[r] + bias3;
  }
}

// ======================= launcher =======================
extern "C" void kernel_launch(void* const* d_in, const int* in_sizes, int n_in,
                              void* d_out, int out_size, void* d_ws, size_t ws_size,
                              hipStream_t stream) {
  (void)in_sizes; (void)n_in; (void)out_size; (void)ws_size;
  const float* x     = (const float*)d_in[0];
  const float* wqkv  = (const float*)d_in[1];
  const float* wproj = (const float*)d_in[2];
  const float* bproj = (const float*)d_in[3];
  const float* rpeq  = (const float*)d_in[4];  // rpe_q_table [H,hd,NB]
  const float* rpek  = (const float*)d_in[5];  // rpe_k_table [H,hd,NB]
  const float* rpev  = (const float*)d_in[6];  // rpe_v_table [H,NB,hd]
  float* y = (float*)d_out;

  // workspace layout (64 MB total)
  char* ws = (char*)d_ws;
  __bf16* Qb  = (__bf16*)(ws);                          // 8 MB  [B,H,N,hd] bf16
  __bf16* Kb  = (__bf16*)(ws + ((size_t)8 << 20));      // 8 MB
  __bf16* Vb  = (__bf16*)(ws + ((size_t)16 << 20));     // 8 MB
  float*  qLw = (float*) (ws + ((size_t)24 << 20));     // 16 MB [B,H,N,64] f32
  float*  kLw = (float*) (ws + ((size_t)40 << 20));     // 16 MB
  __bf16* Ob  = (__bf16*)(ws + ((size_t)56 << 20));     // 8 MB  [B,N,C] bf16

  // 1) QKV projection: 256 row-tiles x 48 col-groups = 12288 waves / 4 per block
  k_qkv<<<3072, 128, 0, stream>>>(x, wqkv, Qb, Kb, Vb);

  // 2) RPE bias panels: B*H*N*64 threads
  k_rpe_qk<<<(BATCH * NH * SEQ * 64) / 256, 256, 0, stream>>>(Qb, Kb, rpeq, rpek,
                                                              qLw, kLw);

  // 3) fused attention: ~296 KB dynamic LDS per block (<= 320 KB/WGP)
  const size_t lds_bytes =
      (size_t)(4 * 16 * SEQ + 8 * 16 * 64) * sizeof(float)   // panels + Ssum + rpeo
      + (32 * HD + 64 * 32) * sizeof(__bf16);                // K stage + V^T stage
  k_attn<<<BATCH * NH * (SEQ / 64), 128, lds_bytes, stream>>>(Qb, Kb, Vb, qLw, kLw,
                                                              rpev, Ob);

  // 4) output projection: 256 row-tiles x 16 col-groups = 4096 waves / 4 per block
  k_proj<<<1024, 128, 0, stream>>>(Ob, wproj, bproj, y);
}
